// BasicBlock_15779709845752
// MI455X (gfx1250) — 8-bit matrix operand order repaired and checked by reading, compiled, not run on hardware
//
#include <hip/hip_runtime.h>
#include <hip/hip_bf16.h>

typedef __attribute__((ext_vector_type(8))) int v8i;
typedef int vint4 __attribute__((vector_size(16)));

#define NB    32
#define CCH   256
#define HH_   56
#define WW_   56
#define HWSZ  3136
#define MTOT  100352
#define KSTEPS 36

#define OFF_APACK 0u
#define OFF_BPACK 25690112u
#define OFF_ALPHA 28049408u
#define OFF_GSUM  28050432u
#define OFF_GSUM2 28051456u
#define OFF_SCSH  28052480u
#define OFF_ZERO  28054528u

#if defined(__has_builtin)
#if __has_builtin(__builtin_amdgcn_global_load_async_to_lds_b128)
#define HAVE_ASYNC_LDS 1
#endif
#endif

static __device__ __forceinline__ v8i mk8(int4 a, int4 b) {
    v8i r;
    r[0]=a.x; r[1]=a.y; r[2]=a.z; r[3]=a.w;
    r[4]=b.x; r[5]=b.y; r[6]=b.z; r[7]=b.w;
    return r;
}

#ifdef HAVE_ASYNC_LDS
static __device__ __forceinline__ void async_slab_load(const signed char* g, signed char* l) {
    auto gp = (__attribute__((address_space(1))) vint4*)(g);
    auto lp = (__attribute__((address_space(3))) vint4*)(l);
    __builtin_amdgcn_global_load_async_to_lds_b128(gp, lp, 0, 0);
    __builtin_amdgcn_global_load_async_to_lds_b128(gp, lp, 16, 0);
}
static __device__ __forceinline__ void wait_async_le2() {
#if __has_builtin(__builtin_amdgcn_s_wait_asynccnt)
    __builtin_amdgcn_s_wait_asynccnt(2);
#else
    asm volatile("s_wait_asynccnt 0x2" ::: "memory");
#endif
}
static __device__ __forceinline__ void wait_async_le0() {
#if __has_builtin(__builtin_amdgcn_s_wait_asynccnt)
    __builtin_amdgcn_s_wait_asynccnt(0);
#else
    asm volatile("s_wait_asynccnt 0x0" ::: "memory");
#endif
}
#endif

__global__ void pack_act(const float* __restrict__ x, signed char* __restrict__ apack) {
    int p = blockIdx.x * 256 + threadIdx.x;
    int n = p / HWSZ;
    int hw = p - n * HWSZ;
    const float* xp = x + (size_t)n * CCH * HWSZ + hw;
    signed char* op = apack + (size_t)p * CCH;
#pragma unroll 4
    for (int cg = 0; cg < 16; ++cg) {
        int d[4];
#pragma unroll
        for (int j = 0; j < 4; ++j) {
            unsigned dw = 0;
#pragma unroll
            for (int b = 0; b < 4; ++b) {
                int c = cg * 16 + j * 4 + b;
                float v = xp[(size_t)c * HWSZ];
                int s = (v > 0.0f) - (v < 0.0f);
                dw |= ((unsigned)(unsigned char)(signed char)s) << (8 * b);
            }
            d[j] = (int)dw;
        }
        int4 o; o.x = d[0]; o.y = d[1]; o.z = d[2]; o.w = d[3];
        *(int4*)(op + cg * 16) = o;
    }
}

__global__ void calc_alpha(const float* __restrict__ wt, float* __restrict__ alpha) {
    __shared__ float red[256];
    int co = blockIdx.x;
    const float* wp = wt + (size_t)co * 2304;
    float s = 0.0f;
    for (int j = threadIdx.x; j < 2304; j += 256) s += fabsf(wp[j]);
    red[threadIdx.x] = s;
    __syncthreads();
    for (int ofs = 128; ofs > 0; ofs >>= 1) {
        if (threadIdx.x < (unsigned)ofs) red[threadIdx.x] += red[threadIdx.x + ofs];
        __syncthreads();
    }
    if (threadIdx.x == 0) alpha[co] = red[0] * (1.0f / 2304.0f);
}

__global__ void pack_wt(const float* __restrict__ wt, int* __restrict__ bpack) {
    int idx = blockIdx.x * 256 + threadIdx.x;
    int lane   = idx & 31;
    int cotile = (idx >> 5) & 15;
    int chunk  = (idx >> 9) & 3;
    int pos    = idx >> 11;
    int co     = cotile * 16 + (lane & 15);
    int khalf  = (lane >> 4) * 16;
    int out[8];
#pragma unroll
    for (int v = 0; v < 8; ++v) {
        int kb = (v & 3) * 4 + (v >> 2) * 32 + khalf;
        unsigned dw = 0;
#pragma unroll
        for (int b = 0; b < 4; ++b) {
            int ci = chunk * 64 + kb + b;
            float wv = wt[(size_t)co * 2304 + (size_t)ci * 9 + pos];
            int s = (wv > 0.0f) - (wv < 0.0f);
            dw |= ((unsigned)(unsigned char)(signed char)s) << (8 * b);
        }
        out[v] = (int)dw;
    }
    int* dst = bpack + (size_t)idx * 8;
    int4 o0; o0.x = out[0]; o0.y = out[1]; o0.z = out[2]; o0.w = out[3];
    int4 o1; o1.x = out[4]; o1.y = out[5]; o1.z = out[6]; o1.w = out[7];
    *(int4*)(dst)     = o0;
    *(int4*)(dst + 4) = o1;
}

__global__ void zero_stats(float* gsum, float* gsum2, int* zbuf) {
    int t = threadIdx.x;
    gsum[t]  = 0.0f;
    gsum2[t] = 0.0f;
    if (t < 16) zbuf[t] = 0;
}

__global__ void __launch_bounds__(256, 2)
conv_wmma(const signed char* __restrict__ apack,
          const int* __restrict__ bpack,
          const float* __restrict__ alpha,
          const signed char* __restrict__ zbuf,
          float* __restrict__ yout,
          float* __restrict__ gsum, float* __restrict__ gsum2) {
#ifdef HAVE_ASYNC_LDS
    __shared__ __align__(16) signed char slab[3][128 * 64];
#else
    __shared__ __align__(16) signed char slab[1][128 * 64];
#endif
    __shared__ float csum[128];
    __shared__ float csum2[128];

    const int t = threadIdx.x;
    const int lane   = t & 31;
    const int laneLo = lane & 15;
    const int laneHi = lane >> 4;
    const int waveId  = t >> 5;
    const int waveCo  = waveId & 3;
    const int wavePix = waveId >> 2;

    const int pixLocal = t >> 1;
    const int half     = t & 1;
    const int ldsOfs   = pixLocal * 64 + half * 32;
    const int p  = blockIdx.x * 128 + pixLocal;
    const int n  = p / HWSZ;
    const int hw = p - n * HWSZ;
    const int h  = hw / WW_;
    const int w  = hw - h * WW_;

    auto srcPtr = [&](int it) -> const signed char* {
        int pos = it >> 2, chunk = it & 3;
        int p3  = pos / 3;
        int dh  = p3 - 1, dww = pos - p3 * 3 - 1;
        int hh = h + dh, ww = w + dww;
        bool valid = (hh >= 0) & (hh < HH_) & (ww >= 0) & (ww < WW_);
        return valid ? apack + ((size_t)(n * HWSZ + hh * WW_ + ww)) * CCH + chunk * 64 + half * 32
                     : zbuf;
    };

    if (t < 128) { csum[t] = 0.0f; csum2[t] = 0.0f; }

    v8i acc[2][4];
#pragma unroll
    for (int mi = 0; mi < 2; ++mi)
#pragma unroll
        for (int nj = 0; nj < 4; ++nj)
#pragma unroll
            for (int r = 0; r < 8; ++r) acc[mi][nj][r] = 0;

#ifdef HAVE_ASYNC_LDS
    async_slab_load(srcPtr(0), &slab[0][ldsOfs]);
    async_slab_load(srcPtr(1), &slab[1][ldsOfs]);
#endif

    int cur = 0;
    for (int it = 0; it < KSTEPS; ++it) {
#ifdef HAVE_ASYNC_LDS
        if (it == KSTEPS - 1) wait_async_le0(); else wait_async_le2();
        __syncthreads();
        if (it + 2 < KSTEPS) {
            int nb = cur + 2; if (nb >= 3) nb -= 3;
            async_slab_load(srcPtr(it + 2), &slab[nb][ldsOfs]);
        }
#else
        __syncthreads();
        const int4* sp = (const int4*)srcPtr(it);
        int4 v0 = sp[0];
        int4 v1 = sp[1];
        *(int4*)&slab[0][ldsOfs]      = v0;
        *(int4*)&slab[0][ldsOfs + 16] = v1;
        __syncthreads();
#endif
        const signed char* sb = slab[cur];

        v8i bfrag[4];
#pragma unroll
        for (int nj = 0; nj < 4; ++nj) {
            int pl = wavePix * 64 + nj * 16 + laneLo;
            const int4* ls = (const int4*)&sb[pl * 64 + laneHi * 16];
            bfrag[nj] = mk8(ls[0], ls[2]);
        }
        v8i afrag[2];
#pragma unroll
        for (int mi = 0; mi < 2; ++mi) {
            int cotile = blockIdx.y * 8 + waveCo * 2 + mi;
            const int4* gp = (const int4*)
                (bpack + ((size_t)((it * 16 + cotile) * 32 + lane)) * 8);
            afrag[mi] = mk8(gp[0], gp[1]);
        }
#pragma unroll
        for (int mi = 0; mi < 2; ++mi)
#pragma unroll
            for (int nj = 0; nj < 4; ++nj)
                acc[mi][nj] = __builtin_amdgcn_wmma_i32_16x16x64_iu8(
                    true, afrag[mi], true, bfrag[nj], acc[mi][nj], false, false);

#ifdef HAVE_ASYNC_LDS
        cur = (cur + 1 == 3) ? 0 : cur + 1;
#endif
    }

#pragma unroll
    for (int mi = 0; mi < 2; ++mi) {
#pragma unroll
        for (int r = 0; r < 8; ++r) {
            const int coLocal = waveCo * 32 + mi * 16 + r + laneHi * 8;
            const int co = blockIdx.y * 128 + coLocal;
            const float al = alpha[co];
#pragma unroll
            for (int nj = 0; nj < 4; ++nj) {
                const int pp = blockIdx.x * 128 + wavePix * 64 + nj * 16 + laneLo;
                const int n2 = pp / HWSZ;
                const int hw2 = pp - n2 * HWSZ;
                const float y = al * (float)acc[mi][nj][r];
                yout[(size_t)(n2 * CCH + co) * HWSZ + hw2] = y;
                atomicAdd(&csum[coLocal], y);
                atomicAdd(&csum2[coLocal], y * y);
            }
        }
    }
    __syncthreads();
    if (t < 128) {
        atomicAdd(&gsum [blockIdx.y * 128 + t], csum [t]);
        atomicAdd(&gsum2[blockIdx.y * 128 + t], csum2[t]);
    }
}

__global__ void finalize_stats(const float* __restrict__ gsum, const float* __restrict__ gsum2,
                               const float* __restrict__ gamma, const float* __restrict__ beta,
                               float* __restrict__ scsh) {
    int c = threadIdx.x;
    const float inv_cnt = 1.0f / (float)MTOT;
    float mu  = gsum[c] * inv_cnt;
    float var = gsum2[c] * inv_cnt - mu * mu;
    float inv = rsqrtf(var + 1e-5f);
    float sc  = gamma[c] * inv;
    scsh[c]       = sc;
    scsh[256 + c] = beta[c] - mu * sc;
}

__global__ void bn_res(const float* __restrict__ x, const float* __restrict__ scsh,
                       float* __restrict__ out) {
    int i = (blockIdx.x * 256 + threadIdx.x) * 4;
    int c = (i / HWSZ) & (CCH - 1);
    float sc = scsh[c], sh = scsh[256 + c];
    float4 y  = *(const float4*)(out + i);
    float4 xv = *(const float4*)(x + i);
    float4 o;
    o.x = fmaf(y.x, sc, sh) + xv.x;
    o.y = fmaf(y.y, sc, sh) + xv.y;
    o.z = fmaf(y.z, sc, sh) + xv.z;
    o.w = fmaf(y.w, sc, sh) + xv.w;
    *(float4*)(out + i) = o;
}

extern "C" void kernel_launch(void* const* d_in, const int* in_sizes, int n_in,
                              void* d_out, int out_size, void* d_ws, size_t ws_size,
                              hipStream_t stream) {
    const float* x     = (const float*)d_in[0];
    const float* wts   = (const float*)d_in[1];
    const float* gamma = (const float*)d_in[2];
    const float* beta  = (const float*)d_in[3];
    float* out = (float*)d_out;

    char* ws = (char*)d_ws;
    signed char* apack = (signed char*)(ws + OFF_APACK);
    int*   bpack = (int*)  (ws + OFF_BPACK);
    float* alpha = (float*)(ws + OFF_ALPHA);
    float* gsum  = (float*)(ws + OFF_GSUM);
    float* gsum2 = (float*)(ws + OFF_GSUM2);
    float* scsh  = (float*)(ws + OFF_SCSH);
    signed char* zbuf = (signed char*)(ws + OFF_ZERO);

    zero_stats<<<1, 256, 0, stream>>>(gsum, gsum2, (int*)zbuf);
    pack_act<<<MTOT / 256, 256, 0, stream>>>(x, apack);
    calc_alpha<<<CCH, 256, 0, stream>>>(wts, alpha);
    pack_wt<<<72, 256, 0, stream>>>(wts, bpack);
    conv_wmma<<<dim3(MTOT / 128, 2), 256, 0, stream>>>(apack, bpack, alpha, zbuf,
                                                       out, gsum, gsum2);
    finalize_stats<<<1, 256, 0, stream>>>(gsum, gsum2, gamma, beta, scsh);
    bn_res<<<(NB * CCH * HWSZ) / (256 * 4), 256, 0, stream>>>(x, scsh, out);
}
